// Mamba_53249004536023
// MI455X (gfx1250) — matrix operand order repaired and checked by reading, not compiled, not run on hardware
//
#include <hip/hip_runtime.h>
#include <hip/hip_bf16.h>

typedef __attribute__((ext_vector_type(16))) _Float16 v16h;
typedef __attribute__((ext_vector_type(8)))  _Float16 v8h;
typedef __attribute__((ext_vector_type(4)))  _Float16 v4h;
typedef __attribute__((ext_vector_type(8)))  float    v8f;
typedef __attribute__((ext_vector_type(4)))  int      v4i;

#define DMODEL 1024
#define DINNER 2048
#define NH     32
#define HD     64
#define DST    128
#define CONVD  2304   // DINNER + 2*DST
#define DPROJ  4384   // 2*DINNER + 2*DST + NH
#define SEQLEN 2048
#define BATCH  2
#define NCHUNK 8
#define CHK    256

#if __has_builtin(__builtin_amdgcn_global_load_async_to_lds_b128) && \
    __has_builtin(__builtin_amdgcn_s_wait_asynccnt)
#define ASYNC_OK 1
#endif

#define GAS __attribute__((address_space(1)))
#define LAS __attribute__((address_space(3)))

__device__ __forceinline__ float siluf(float v){ return v / (1.f + __expf(-v)); }

// 16-byte global -> LDS copy; async (ASYNCcnt) on CDNA5, sync fallback.
__device__ __forceinline__ void cp16(const _Float16* g, _Float16* l){
#ifdef ASYNC_OK
  __builtin_amdgcn_global_load_async_to_lds_b128((GAS v4i*)g, (LAS v4i*)l, 0, 0);
#else
  *(v8h*)l = *(const v8h*)g;
#endif
}
template<int N>
__device__ __forceinline__ void wait_async(){
#ifdef ASYNC_OK
  __builtin_amdgcn_s_wait_asynccnt((unsigned short)N);
#endif
}

// A-fragment K index: contiguous order k = 16*hi + i (matches B fills)
__device__ __forceinline__ int a_k(int i, int hi){
  return hi * 16 + i;
}
__device__ __forceinline__ void vzero8(v8f& a){
#pragma unroll
  for (int i = 0; i < 8; ++i) a[i] = 0.f;
}

// =====================================================================
// f32 -> f16 cast (vectorized); n must be a multiple of 4
// =====================================================================
__global__ __launch_bounds__(256)
void k_cast_f16(const float* __restrict__ in, _Float16* __restrict__ out, long n)
{
  long i = ((long)blockIdx.x * 256 + threadIdx.x) * 4;
  if (i + 4 <= n) {
    const float4 v = *(const float4*)(in + i);
    v4h o;
    o[0] = (_Float16)v.x; o[1] = (_Float16)v.y;
    o[2] = (_Float16)v.z; o[3] = (_Float16)v.w;
    *(v4h*)(out + i) = o;
  }
}

// =====================================================================
// WMMA GEMM: C[M,N] = A[M,K] * B[N,K]^T  (f16 in, f32 out, optional f16 out)
// 256 thr = 8 waves (4x2), tile 128x64, K step 32, double-buffered async LDS
// =====================================================================
__global__ __launch_bounds__(256)
void k_gemm_wmma(const _Float16* __restrict__ A, const _Float16* __restrict__ Bw,
                 float* __restrict__ C, _Float16* __restrict__ Ch,
                 int M, int N, int K)
{
  __shared__ _Float16 As[2][128][40];   // 40 halves = 80B row stride (16B mult)
  __shared__ _Float16 Bs[2][64][40];
  const int tid  = threadIdx.x;
  const int lane = tid & 31;
  const int wave = tid >> 5;
  const int hi   = lane >> 4;
  const int l16  = lane & 15;
  const int wm   = (wave >> 1) * 32;
  const int wn   = (wave & 1) * 32;
  const int bm   = blockIdx.y * 128;
  const int bn   = blockIdx.x * 64;

  auto stage = [&](int buf, int k0){
    // A tile 128x32 halves = 512 16B chunks -> 2/thread
#pragma unroll
    for (int cc = 0; cc < 2; ++cc) {
      int ch = tid + cc * 256;
      int r = ch >> 2, c8 = (ch & 3) * 8;
      cp16(&A[(size_t)(bm + r) * K + k0 + c8], &As[buf][r][c8]);
    }
    // B tile 64x32 halves = 256 chunks -> 1/thread (clamp row at N edge)
    {
      int r = tid >> 2, c8 = (tid & 3) * 8;
      int ng = bn + r; if (ng > N - 1) ng = N - 1;
      cp16(&Bw[(size_t)ng * K + k0 + c8], &Bs[buf][r][c8]);
    }
  };

  v8f acc[2][2];
#pragma unroll
  for (int mi = 0; mi < 2; ++mi)
#pragma unroll
    for (int ni = 0; ni < 2; ++ni) vzero8(acc[mi][ni]);

  const int nk = K / 32;
  stage(0, 0);
  for (int it = 0; it < nk; ++it) {
    const int cur = it & 1;
    if (it + 1 < nk) { stage(cur ^ 1, (it + 1) * 32); wait_async<3>(); }
    else             { wait_async<0>(); }
    __syncthreads();

    v16h af[2], bf[2];
#pragma unroll
    for (int mi = 0; mi < 2; ++mi) {
      int mrow = wm + mi * 16 + l16;
#pragma unroll
      for (int i = 0; i < 16; ++i) af[mi][i] = As[cur][mrow][a_k(i, hi)];
    }
#pragma unroll
    for (int ni = 0; ni < 2; ++ni) {
      int nc = wn + ni * 16 + l16;
#pragma unroll
      for (int i = 0; i < 16; ++i) bf[ni][i] = Bs[cur][nc][hi * 16 + i];
    }
#pragma unroll
    for (int mi = 0; mi < 2; ++mi)
#pragma unroll
      for (int ni = 0; ni < 2; ++ni)
        acc[mi][ni] = __builtin_amdgcn_wmma_f32_16x16x32_f16(
            false, af[mi], false, bf[ni], (short)0, acc[mi][ni], false, false);
    __syncthreads();   // protects ping-pong WAR (both paths)
  }

#pragma unroll
  for (int mi = 0; mi < 2; ++mi)
#pragma unroll
    for (int ni = 0; ni < 2; ++ni)
#pragma unroll
      for (int r = 0; r < 8; ++r) {
        int row = bm + wm + mi * 16 + hi * 8 + r;
        int col = bn + wn + ni * 16 + l16;
        if (row < M && col < N) {
          float v = acc[mi][ni][r];
          C[(size_t)row * N + col] = v;
          if (Ch) Ch[(size_t)row * N + col] = (_Float16)v;
        }
      }
}

// =====================================================================
// dt = softplus(zx[..., -NH:] + dt_bias)
// =====================================================================
__global__ __launch_bounds__(256)
void k_dt(const float* __restrict__ zx, const float* __restrict__ bias,
          float* __restrict__ dtv)
{
  int idx = blockIdx.x * 256 + threadIdx.x;
  if (idx >= BATCH * SEQLEN * NH) return;
  int h = idx & (NH - 1);
  int row = idx >> 5;
  float u = zx[(size_t)row * DPROJ + (DPROJ - NH) + h] + bias[h];
  dtv[idx] = (u > 20.f) ? u : log1pf(__expf(u));
}

// =====================================================================
// depthwise causal conv (k=4) + bias + SiLU; writes f16
// =====================================================================
__global__ __launch_bounds__(256)
void k_conv_silu(const float* __restrict__ zx, const float* __restrict__ cw,
                 const float* __restrict__ cb, _Float16* __restrict__ xBCh)
{
  size_t idx = (size_t)blockIdx.x * 256 + threadIdx.x;
  if (idx >= (size_t)BATCH * SEQLEN * CONVD) return;
  int ch = (int)(idx % CONVD);
  int l  = (int)((idx / CONVD) % SEQLEN);
  int b  = (int)(idx / ((size_t)CONVD * SEQLEN));
  float acc = cb[ch];
#pragma unroll
  for (int t = 0; t < 4; ++t) {
    int ls = l + t - 3;
    if (ls >= 0)
      acc += cw[ch * 4 + t] * zx[(size_t)(b * SEQLEN + ls) * DPROJ + DINNER + ch];
  }
  xBCh[idx] = (_Float16)siluf(acc);
}

// =====================================================================
// SSD states: states[p,n] = sum_l B[l,n]*decay_l*(x*dt)[l,p]
// M=128(n) x N=64(p), K=256(l); 8 waves 4x2
// =====================================================================
__global__ __launch_bounds__(256)
void k_ssd_states(const _Float16* __restrict__ xBCh, const float* __restrict__ dtv,
                  const float* __restrict__ Alog, float* __restrict__ states,
                  float* __restrict__ csum)
{
  __shared__ _Float16 sB[CHK][136];    // 272B stride
  __shared__ _Float16 sXd[CHK][72];    // 144B stride
  __shared__ float sAcs[CHK];
  __shared__ float sdt[CHK];
  const int h = blockIdx.x, c = blockIdx.y, b = blockIdx.z;
  const int tid = threadIdx.x, lane = tid & 31, wave = tid >> 5;
  const int hi = lane >> 4, l16 = lane & 15;
  const int rowbase = b * SEQLEN + c * CHK;
  const float A = -__expf(Alog[h]);

  { float d = dtv[(size_t)(rowbase + tid) * NH + h]; sdt[tid] = d; sAcs[tid] = d * A; }

  // kick off B staging early (async, overlaps the scan)
#pragma unroll
  for (int i = 0; i < 16; ++i) {
    int ch = tid + i * 256;
    int l = ch >> 4, c8 = (ch & 15) * 8;
    cp16(&xBCh[(size_t)(rowbase + l) * CONVD + DINNER + c8], &sB[l][c8]);
  }

  __syncthreads();
  for (int off = 1; off < CHK; off <<= 1) {     // inclusive cumsum of dA
    float v = (tid >= off) ? sAcs[tid - off] : 0.f;
    __syncthreads();
    sAcs[tid] += v;
    __syncthreads();
  }
  const float Atot = sAcs[CHK - 1];

  // x*dt*decay staging (register path, vectorized)
#pragma unroll
  for (int i = 0; i < 8; ++i) {
    int ch = tid + i * 256;
    int l = ch >> 3, c8 = (ch & 7) * 8;
    v8h x = *(const v8h*)&xBCh[(size_t)(rowbase + l) * CONVD + h * HD + c8];
    float sc = sdt[l] * __expf(Atot - sAcs[l]);
    v8h o;
#pragma unroll
    for (int e = 0; e < 8; ++e) o[e] = (_Float16)((float)x[e] * sc);
    *(v8h*)&sXd[l][c8] = o;
  }
  wait_async<0>();
  __syncthreads();

  const int wn = (wave >> 1) * 32, wp = (wave & 1) * 32;
  v8f acc[2][2];
#pragma unroll
  for (int mi = 0; mi < 2; ++mi)
#pragma unroll
    for (int ni = 0; ni < 2; ++ni) vzero8(acc[mi][ni]);

  for (int kk = 0; kk < 8; ++kk) {
    int kb = kk * 32;
    v16h af[2], bf[2];
#pragma unroll
    for (int mi = 0; mi < 2; ++mi) {
      int mrow = wn + mi * 16 + l16;
#pragma unroll
      for (int i = 0; i < 16; ++i) af[mi][i] = sB[kb + a_k(i, hi)][mrow];
    }
#pragma unroll
    for (int ni = 0; ni < 2; ++ni) {
      int nc = wp + ni * 16 + l16;
#pragma unroll
      for (int i = 0; i < 16; ++i) bf[ni][i] = sXd[kb + hi * 16 + i][nc];
    }
#pragma unroll
    for (int mi = 0; mi < 2; ++mi)
#pragma unroll
      for (int ni = 0; ni < 2; ++ni)
        acc[mi][ni] = __builtin_amdgcn_wmma_f32_16x16x32_f16(
            false, af[mi], false, bf[ni], (short)0, acc[mi][ni], false, false);
  }

  size_t sbase = (size_t)((b * NCHUNK + c) * NH + h) * HD * DST;
#pragma unroll
  for (int mi = 0; mi < 2; ++mi)
#pragma unroll
    for (int ni = 0; ni < 2; ++ni)
#pragma unroll
      for (int r = 0; r < 8; ++r) {
        int n = wn + mi * 16 + hi * 8 + r;
        int p = wp + ni * 16 + l16;
        states[sbase + (size_t)p * DST + n] = acc[mi][ni][r];
      }
  if (tid == 0) csum[(size_t)(b * NH + h) * NCHUNK + c] = Atot;
}

// =====================================================================
// inter-chunk recurrence
// =====================================================================
__global__ __launch_bounds__(256)
void k_interchunk(const float* __restrict__ states, const float* __restrict__ csum,
                  float* __restrict__ prevb)
{
  int bh = blockIdx.x;
  int b = bh >> 5, h = bh & 31;
  float cs[NCHUNK];
#pragma unroll
  for (int c = 0; c < NCHUNK; ++c) cs[c] = __expf(csum[(size_t)bh * NCHUNK + c]);
  for (int e = threadIdx.x; e < HD * DST; e += 256) {
    float carry = 0.f;
#pragma unroll
    for (int c = 0; c < NCHUNK; ++c) {
      size_t idx = (size_t)((b * NCHUNK + c) * NH + h) * HD * DST + e;
      prevb[idx] = carry;
      carry = carry * cs[c] + states[idx];
    }
  }
}

// =====================================================================
// SSD Y: Y = exp(Acs_l)*(C@prev) + sum_tiles (mask.decay o C.B^T) @ (x*dt) + D*x
// =====================================================================
__global__ __launch_bounds__(256)
void k_ssd_y(const _Float16* __restrict__ xBCh, const float* __restrict__ dtv,
             const float* __restrict__ Alog, const float* __restrict__ Dpar,
             const float* __restrict__ prevb, float* __restrict__ ybuf)
{
  __shared__ _Float16 sC[CHK][136];
  __shared__ _Float16 sBt[64][136];
  __shared__ _Float16 sXt[64][72];
  __shared__ _Float16 sG[CHK][72];
  __shared__ _Float16 sPrev[HD][136];   // [p][n]
  __shared__ float sAcs[CHK];
  __shared__ float sdt[CHK];
  const int h = blockIdx.x, c = blockIdx.y, b = blockIdx.z;
  const int tid = threadIdx.x, lane = tid & 31, wave = tid >> 5;
  const int hi = lane >> 4, l16 = lane & 15;
  const int rowbase = b * SEQLEN + c * CHK;
  const float A = -__expf(Alog[h]);

  { float d = dtv[(size_t)(rowbase + tid) * NH + h]; sdt[tid] = d; sAcs[tid] = d * A; }

  // C tile staging early (async, overlaps scan)
#pragma unroll
  for (int i = 0; i < 16; ++i) {
    int ch = tid + i * 256;
    int l = ch >> 4, c8 = (ch & 15) * 8;
    cp16(&xBCh[(size_t)(rowbase + l) * CONVD + DINNER + DST + c8], &sC[l][c8]);
  }

  __syncthreads();
  for (int off = 1; off < CHK; off <<= 1) {
    float v = (tid >= off) ? sAcs[tid - off] : 0.f;
    __syncthreads();
    sAcs[tid] += v;
    __syncthreads();
  }

  // prev (f32 -> f16), stored [p][n]
  {
    size_t pb = (size_t)((b * NCHUNK + c) * NH + h) * HD * DST;
#pragma unroll
    for (int i = 0; i < 8; ++i) {
      long e0 = (long)(tid + i * 256) * 4;
      int p = (int)(e0 >> 7), n = (int)(e0 & 127);
      const float4 v = *(const float4*)&prevb[pb + e0];
      v4h o;
      o[0] = (_Float16)v.x; o[1] = (_Float16)v.y;
      o[2] = (_Float16)v.z; o[3] = (_Float16)v.w;
      *(v4h*)&sPrev[p][n] = o;
    }
  }
  wait_async<0>();
  __syncthreads();

  const int rb = wave * 32;
  v8f yacc[2][4];

  // ---- Y_off = exp(Acs_l) * (C @ prev) ----
  {
    v8f offa[2][4];
#pragma unroll
    for (int mi = 0; mi < 2; ++mi)
#pragma unroll
      for (int ni = 0; ni < 4; ++ni) vzero8(offa[mi][ni]);
    for (int kk = 0; kk < 4; ++kk) {
      int kb = kk * 32;
      v16h af[2], bf[4];
#pragma unroll
      for (int mi = 0; mi < 2; ++mi) {
        int mrow = rb + mi * 16 + l16;
#pragma unroll
        for (int i = 0; i < 16; ++i) af[mi][i] = sC[mrow][kb + a_k(i, hi)];
      }
#pragma unroll
      for (int ni = 0; ni < 4; ++ni) {
        int nc = ni * 16 + l16;
#pragma unroll
        for (int i = 0; i < 16; ++i) bf[ni][i] = sPrev[nc][kb + hi * 16 + i];
      }
#pragma unroll
      for (int mi = 0; mi < 2; ++mi)
#pragma unroll
        for (int ni = 0; ni < 4; ++ni)
          offa[mi][ni] = __builtin_amdgcn_wmma_f32_16x16x32_f16(
              false, af[mi], false, bf[ni], (short)0, offa[mi][ni], false, false);
    }
#pragma unroll
    for (int mi = 0; mi < 2; ++mi)
#pragma unroll
      for (int ni = 0; ni < 4; ++ni)
#pragma unroll
        for (int r = 0; r < 8; ++r) {
          int l = rb + mi * 16 + hi * 8 + r;
          yacc[mi][ni][r] = offa[mi][ni][r] * __expf(sAcs[l]);
        }
  }

  // ---- flash-style s-tiles ----
  for (int st = 0; st < 4; ++st) {
    __syncthreads();
    // B tile: 64x128 halves = 1024 chunks -> 4/thread (async)
#pragma unroll
    for (int i = 0; i < 4; ++i) {
      int ch = tid + i * 256;
      int s = ch >> 4, c8 = (ch & 15) * 8;
      cp16(&xBCh[(size_t)(rowbase + st * 64 + s) * CONVD + DINNER + c8], &sBt[s][c8]);
    }
    // x*dt tile (register path)
#pragma unroll
    for (int i = 0; i < 2; ++i) {
      int ch = tid + i * 256;
      int s = ch >> 3, c8 = (ch & 7) * 8;
      v8h x = *(const v8h*)&xBCh[(size_t)(rowbase + st * 64 + s) * CONVD + h * HD + c8];
      float sc = sdt[st * 64 + s];
      v8h o;
#pragma unroll
      for (int e = 0; e < 8; ++e) o[e] = (_Float16)((float)x[e] * sc);
      *(v8h*)&sXt[s][c8] = o;
    }
    wait_async<0>();
    __syncthreads();

    v8f g[2][4];
#pragma unroll
    for (int mi = 0; mi < 2; ++mi)
#pragma unroll
      for (int ni = 0; ni < 4; ++ni) vzero8(g[mi][ni]);
    for (int kk = 0; kk < 4; ++kk) {          // G = C @ B^T over n
      int kb = kk * 32;
      v16h af[2], bf[4];
#pragma unroll
      for (int mi = 0; mi < 2; ++mi) {
        int mrow = rb + mi * 16 + l16;
#pragma unroll
        for (int i = 0; i < 16; ++i) af[mi][i] = sC[mrow][kb + a_k(i, hi)];
      }
#pragma unroll
      for (int ni = 0; ni < 4; ++ni) {
        int nc = ni * 16 + l16;
#pragma unroll
        for (int i = 0; i < 16; ++i) bf[ni][i] = sBt[nc][kb + hi * 16 + i];
      }
#pragma unroll
      for (int mi = 0; mi < 2; ++mi)
#pragma unroll
        for (int ni = 0; ni < 4; ++ni)
          g[mi][ni] = __builtin_amdgcn_wmma_f32_16x16x32_f16(
              false, af[mi], false, bf[ni], (short)0, g[mi][ni], false, false);
    }
    // causal mask + decay -> sG (each wave owns its 32 rows)
#pragma unroll
    for (int mi = 0; mi < 2; ++mi)
#pragma unroll
      for (int ni = 0; ni < 4; ++ni)
#pragma unroll
        for (int r = 0; r < 8; ++r) {
          int l = rb + mi * 16 + hi * 8 + r;
          int sg = st * 64 + ni * 16 + l16;
          float v = (sg <= l) ? g[mi][ni][r] * __expf(sAcs[l] - sAcs[sg]) : 0.f;
          sG[l][ni * 16 + l16] = (_Float16)v;
        }
    __syncthreads();

    for (int kk = 0; kk < 2; ++kk) {          // Y += G @ (x*dt) over s
      int kb = kk * 32;
      v16h af[2], bf[4];
#pragma unroll
      for (int mi = 0; mi < 2; ++mi) {
        int mrow = rb + mi * 16 + l16;
#pragma unroll
        for (int i = 0; i < 16; ++i) af[mi][i] = sG[mrow][kb + a_k(i, hi)];
      }
#pragma unroll
      for (int ni = 0; ni < 4; ++ni) {
        int nc = ni * 16 + l16;
#pragma unroll
        for (int i = 0; i < 16; ++i) bf[ni][i] = sXt[kb + hi * 16 + i][nc];
      }
#pragma unroll
      for (int mi = 0; mi < 2; ++mi)
#pragma unroll
        for (int ni = 0; ni < 4; ++ni)
          yacc[mi][ni] = __builtin_amdgcn_wmma_f32_16x16x32_f16(
              false, af[mi], false, bf[ni], (short)0, yacc[mi][ni], false, false);
    }
  }

  const float Dh = Dpar[h];
#pragma unroll
  for (int mi = 0; mi < 2; ++mi)
#pragma unroll
    for (int ni = 0; ni < 4; ++ni)
#pragma unroll
      for (int r = 0; r < 8; ++r) {
        int l = rb + mi * 16 + hi * 8 + r;
        int p = ni * 16 + l16;
        float xs = (float)xBCh[(size_t)(rowbase + l) * CONVD + h * HD + p];
        ybuf[(size_t)(rowbase + l) * DINNER + h * HD + p] = yacc[mi][ni][r] + Dh * xs;
      }
}

// =====================================================================
// y = y * silu(z); RMSNorm with norm_w; writes f16 for out_proj
// =====================================================================
__global__ __launch_bounds__(256)
void k_gate_norm(const float* __restrict__ y, const float* __restrict__ zx,
                 const float* __restrict__ nw, _Float16* __restrict__ yh)
{
  __shared__ float red[256];
  int row = blockIdx.x, tid = threadIdx.x;
  size_t by = (size_t)row * DINNER, bz = (size_t)row * DPROJ;
  float lv[8]; float s = 0.f;
#pragma unroll
  for (int i = 0; i < 8; ++i) {
    int e = tid + i * 256;
    float z = zx[bz + e];
    float v = y[by + e] * siluf(z);
    lv[i] = v; s += v * v;
  }
  red[tid] = s; __syncthreads();
  for (int o = 128; o > 0; o >>= 1) {
    if (tid < o) red[tid] += red[tid + o];
    __syncthreads();
  }
  float scale = rsqrtf(red[0] / (float)DINNER + 1e-5f);
#pragma unroll
  for (int i = 0; i < 8; ++i) {
    int e = tid + i * 256;
    yh[by + e] = (_Float16)(lv[i] * scale * nw[e]);
  }
}

// =====================================================================
// final LayerNorm over DMODEL
// =====================================================================
__global__ __launch_bounds__(256)
void k_final_ln(const float* __restrict__ x, const float* __restrict__ w,
                const float* __restrict__ bb, float* __restrict__ out)
{
  __shared__ float r1[256], r2[256];
  int row = blockIdx.x, tid = threadIdx.x;
  size_t base = (size_t)row * DMODEL;
  float lv[4]; float s1 = 0.f, s2 = 0.f;
#pragma unroll
  for (int i = 0; i < 4; ++i) {
    int e = tid + i * 256;
    float v = x[base + e];
    lv[i] = v; s1 += v; s2 += v * v;
  }
  r1[tid] = s1; r2[tid] = s2; __syncthreads();
  for (int o = 128; o > 0; o >>= 1) {
    if (tid < o) { r1[tid] += r1[tid + o]; r2[tid] += r2[tid + o]; }
    __syncthreads();
  }
  float mu  = r1[0] / (float)DMODEL;
  float var = r2[0] / (float)DMODEL - mu * mu;
  float sc  = rsqrtf(var + 1e-5f);
#pragma unroll
  for (int i = 0; i < 4; ++i) {
    int e = tid + i * 256;
    out[base + e] = (lv[i] - mu) * sc * w[e] + bb[e];
  }
}

// =====================================================================
extern "C" void kernel_launch(void* const* d_in, const int* in_sizes, int n_in,
                              void* d_out, int out_size, void* d_ws, size_t ws_size,
                              hipStream_t stream)
{
  (void)in_sizes; (void)n_in; (void)out_size; (void)ws_size;
  const float* seq   = (const float*)d_in[0];
  const float* W_in  = (const float*)d_in[1];
  const float* convw = (const float*)d_in[2];
  const float* convb = (const float*)d_in[3];
  const float* dtb   = (const float*)d_in[4];
  const float* Alog  = (const float*)d_in[5];
  const float* Dpar  = (const float*)d_in[6];
  const float* normw = (const float*)d_in[7];
  const float* W_out = (const float*)d_in[8];
  const float* lnw   = (const float*)d_in[9];
  const float* lnb   = (const float*)d_in[10];

  float* ws = (float*)d_ws;
  size_t off = 0;
  float* zx     = ws + off; off += (size_t)BATCH * SEQLEN * DPROJ;
  float* dtv    = ws + off; off += (size_t)BATCH * SEQLEN * NH;
  float* states = ws + off; off += (size_t)BATCH * NCHUNK * NH * HD * DST;
  float* prevb  = ws + off; off += (size_t)BATCH * NCHUNK * NH * HD * DST;
  float* csum   = ws + off; off += (size_t)BATCH * NH * NCHUNK;
  float* ybuf   = ws + off; off += (size_t)BATCH * SEQLEN * DINNER;
  float* xf     = ws + off; off += (size_t)BATCH * SEQLEN * DMODEL;
  // f16 regions (counted in float slots = halves/2; all offsets stay 16B-aligned)
  _Float16* xh    = (_Float16*)(ws + off); off += (size_t)BATCH * SEQLEN * DMODEL / 2;
  _Float16* yh    = (_Float16*)(ws + off); off += (size_t)BATCH * SEQLEN * DINNER / 2;
  _Float16* xBCh  = (_Float16*)(ws + off); off += (size_t)BATCH * SEQLEN * CONVD / 2;
  _Float16* WinH  = (_Float16*)(ws + off); off += (size_t)4 * DPROJ * DMODEL / 2;
  _Float16* WoutH = (_Float16*)(ws + off); off += (size_t)4 * DMODEL * DINNER / 2;

  const int Mrows = BATCH * SEQLEN;

  // one-time f16 conversions
  {
    long nWin = (long)4 * DPROJ * DMODEL;
    long nWout = (long)4 * DMODEL * DINNER;
    long nX = (long)Mrows * DMODEL;
    k_cast_f16<<<(int)(nWin / 4 / 256), 256, 0, stream>>>(W_in, WinH, nWin);
    k_cast_f16<<<(int)(nWout / 4 / 256), 256, 0, stream>>>(W_out, WoutH, nWout);
    k_cast_f16<<<(int)(nX / 4 / 256), 256, 0, stream>>>(seq, xh, nX);
  }

  for (int layer = 0; layer < 4; ++layer) {
    k_gemm_wmma<<<dim3((DPROJ + 63) / 64, (Mrows + 127) / 128), 256, 0, stream>>>(
        xh, WinH + (size_t)layer * DPROJ * DMODEL, zx, ((_Float16*)nullptr),
        Mrows, DPROJ, DMODEL);

    k_dt<<<(Mrows * NH + 255) / 256, 256, 0, stream>>>(zx, dtb + layer * NH, dtv);

    k_conv_silu<<<(int)(((size_t)Mrows * CONVD + 255) / 256), 256, 0, stream>>>(
        zx, convw + (size_t)layer * CONVD * 4, convb + (size_t)layer * CONVD, xBCh);

    k_ssd_states<<<dim3(NH, NCHUNK, BATCH), 256, 0, stream>>>(
        xBCh, dtv, Alog + layer * NH, states, csum);

    k_interchunk<<<BATCH * NH, 256, 0, stream>>>(states, csum, prevb);

    k_ssd_y<<<dim3(NH, NCHUNK, BATCH), 256, 0, stream>>>(
        xBCh, dtv, Alog + layer * NH, Dpar + layer * NH, prevb, ybuf);

    k_gate_norm<<<Mrows, 256, 0, stream>>>(ybuf, zx, normw + (size_t)layer * DINNER, yh);

    // out_proj writes f32 (for final LN) and f16 (next layer's in_proj input)
    k_gemm_wmma<<<dim3((DMODEL + 63) / 64, (Mrows + 127) / 128), 256, 0, stream>>>(
        yh, WoutH + (size_t)layer * DMODEL * DINNER, xf, xh, Mrows, DMODEL, DINNER);
  }
  k_final_ln<<<Mrows, 256, 0, stream>>>(xf, lnw, lnb, (float*)d_out);
}
